// CrossModalAttention_48902497632793
// MI455X (gfx1250) — matrix operand order repaired and checked by reading, not compiled, not run on hardware
//
#include <hip/hip_runtime.h>
#include <hip/hip_bf16.h>

// ---------------------------------------------------------------------------
// CrossModalAttention for MI455X (gfx1250), wave32 + v_wmma_f32_16x16x32_f16.
// Fixed shapes: B=2, Tq=Tc=2048, dm=dq=dc=1024, H=16, hd=64.
// ---------------------------------------------------------------------------

typedef __attribute__((ext_vector_type(16))) _Float16 v16h;
typedef __attribute__((ext_vector_type(8)))  _Float16 v8h;
typedef __attribute__((ext_vector_type(8)))  float    v8f;

#define EPSV 1e-6f
constexpr int BATCH = 2;
constexpr int T     = 2048;
constexpr int DMODEL= 1024;
constexpr int NHEAD = 16;
constexpr int HD    = 64;
constexpr int MROWS = BATCH * T;     // 4096 rows for all projections

union F8  { v8f  v; float    f[8]; };
union H16 { v16h v; v8h      h[2]; };

// A-operand fragment (16x32 f16): lane r = row M, K = 16h..16h+15 (contiguous)
__device__ inline v16h load_frag_a(const _Float16* __restrict__ p, int hf) {
  H16 u;
  u.h[0] = *(const v8h*)(p + 16 * hf);
  u.h[1] = *(const v8h*)(p + 16 * hf + 8);
  return u.v;
}
// B-operand fragment (32x16 f16) from B^T rows: lane r = column N,
// half-wave 0 holds K 0..15, half-wave 1 holds K 16..31 (contiguous 32B).
__device__ inline v16h load_frag_b(const _Float16* __restrict__ p, int hf) {
  return *(const v16h*)(p + 16 * hf);
}

__device__ inline float wave16_max(float x) {
  #pragma unroll
  for (int off = 1; off < 16; off <<= 1) x = fmaxf(x, __shfl_xor(x, off, 32));
  return x;
}
__device__ inline float wave16_sum(float x) {
  #pragma unroll
  for (int off = 1; off < 16; off <<= 1) x += __shfl_xor(x, off, 32);
  return x;
}

// ---------------------------------------------------------------------------
// rmsnorm(x, g) -> f16, one block per row of 1024
// ---------------------------------------------------------------------------
__global__ void rmsnorm_cast_kernel(const float* __restrict__ x,
                                    const float* __restrict__ g,
                                    _Float16* __restrict__ y) {
  const int row = blockIdx.x;
  const float* xr = x + (size_t)row * DMODEL;
  const int tid = threadIdx.x;
  float ss = 0.f;
  for (int i = tid; i < DMODEL; i += 256) { float v = xr[i]; ss += v * v; }
  #pragma unroll
  for (int off = 16; off > 0; off >>= 1) ss += __shfl_xor(ss, off, 32);
  __shared__ float red[8];
  if ((tid & 31) == 0) red[tid >> 5] = ss;
  __syncthreads();
  float tot = red[0] + red[1] + red[2] + red[3] + red[4] + red[5] + red[6] + red[7];
  float rinv = rsqrtf(tot * (1.0f / DMODEL) + EPSV);
  _Float16* yr = y + (size_t)row * DMODEL;
  for (int i = tid; i < DMODEL; i += 256) yr[i] = (_Float16)(xr[i] * rinv * g[i]);
}

// ---------------------------------------------------------------------------
// W[K][N] f32 -> Wt[N][K] f16 (32x32 LDS tiles)
// ---------------------------------------------------------------------------
__global__ void wtrans_kernel(const float* __restrict__ W, _Float16* __restrict__ Wt) {
  __shared__ float tile[32][33];
  const int n0 = blockIdx.x * 32, k0 = blockIdx.y * 32;
  const int tx = threadIdx.x & 31, ty = threadIdx.x >> 5;
  for (int i = ty; i < 32; i += 8)
    tile[i][tx] = W[(size_t)(k0 + i) * DMODEL + (n0 + tx)];
  __syncthreads();
  for (int i = ty; i < 32; i += 8)
    Wt[(size_t)(n0 + i) * DMODEL + (k0 + tx)] = (_Float16)tile[tx][i];
}

// ---------------------------------------------------------------------------
// C = A[4096x1024] @ Bt^T, WMMA f16. 8 waves/block, wave tile 32x32,
// block tile 128(M) x 64(N). mode: 0 -> f16 per-head [b][h][t][hd]
//                             1 -> f16 per-head transposed [b][h][hd][t]
//                             2 -> f32 plain [m][n]
// ---------------------------------------------------------------------------
__global__ void gemm_wmma_kernel(const _Float16* __restrict__ A,
                                 const _Float16* __restrict__ Bt,
                                 _Float16* __restrict__ outH,
                                 float* __restrict__ outF,
                                 int mode) {
  const int tid  = threadIdx.x;
  const int wid  = tid >> 5, lane = tid & 31;
  const int r    = lane & 15, hf = lane >> 4;
  const int mw   = wid >> 1, nw = wid & 1;
  const int m0   = blockIdx.y * 128 + mw * 32;
  const int n0   = blockIdx.x * 64 + nw * 32;

  F8 c[2][2];
  #pragma unroll
  for (int mt = 0; mt < 2; mt++)
    #pragma unroll
    for (int nt = 0; nt < 2; nt++)
      #pragma unroll
      for (int i = 0; i < 8; i++) c[mt][nt].f[i] = 0.f;

  const _Float16* a0p = A  + (size_t)(m0 + 0 * 16 + r) * DMODEL;
  const _Float16* a1p = A  + (size_t)(m0 + 1 * 16 + r) * DMODEL;
  const _Float16* b0p = Bt + (size_t)(n0 + 0 * 16 + r) * DMODEL;
  const _Float16* b1p = Bt + (size_t)(n0 + 1 * 16 + r) * DMODEL;

  for (int kk = 0; kk < DMODEL; kk += 32) {
    v16h a0 = load_frag_a(a0p + kk, hf);
    v16h a1 = load_frag_a(a1p + kk, hf);
    v16h b0 = load_frag_b(b0p + kk, hf);
    v16h b1 = load_frag_b(b1p + kk, hf);
    c[0][0].v = __builtin_amdgcn_wmma_f32_16x16x32_f16(false, a0, false, b0, (short)0, c[0][0].v, false, false);
    c[0][1].v = __builtin_amdgcn_wmma_f32_16x16x32_f16(false, a0, false, b1, (short)0, c[0][1].v, false, false);
    c[1][0].v = __builtin_amdgcn_wmma_f32_16x16x32_f16(false, a1, false, b0, (short)0, c[1][0].v, false, false);
    c[1][1].v = __builtin_amdgcn_wmma_f32_16x16x32_f16(false, a1, false, b1, (short)0, c[1][1].v, false, false);
  }

  #pragma unroll
  for (int mt = 0; mt < 2; mt++)
    #pragma unroll
    for (int nt = 0; nt < 2; nt++)
      #pragma unroll
      for (int i = 0; i < 8; i++) {
        const int m = m0 + mt * 16 + i + 8 * hf;
        const int n = n0 + nt * 16 + r;
        const float v = c[mt][nt].f[i];
        if (mode == 0) {
          const int bb = m >> 11, t = m & (T - 1), hh = n >> 6, d = n & (HD - 1);
          outH[(((size_t)(bb * NHEAD + hh) * T + t) * HD) + d] = (_Float16)v;
        } else if (mode == 1) {
          const int bb = m >> 11, t = m & (T - 1), hh = n >> 6, d = n & (HD - 1);
          outH[(((size_t)(bb * NHEAD + hh) * HD + d) * T) + t] = (_Float16)v;
        } else {
          outF[(size_t)m * DMODEL + n] = v;
        }
      }
}

// ---------------------------------------------------------------------------
// Flash attention: one wave = one 16-row q tile of one (b, head).
// Q,K: [b][h][t][hd] f16, Vt: [b][h][hd][t] f16. out: [b*t][dm] f16.
// ---------------------------------------------------------------------------
__global__ void flash_attn_kernel(const _Float16* __restrict__ Qb,
                                  const _Float16* __restrict__ Kb,
                                  const _Float16* __restrict__ Vtb,
                                  const unsigned char* __restrict__ mask,
                                  _Float16* __restrict__ out) {
  __shared__ __align__(16) _Float16 pbuf[8 * 512];  // 16x32 f16 tile per wave
  const int tid  = threadIdx.x;
  const int wid  = tid >> 5, lane = tid & 31;
  const int r    = lane & 15, hf = lane >> 4;

  const int task = blockIdx.x * 8 + wid;           // 4096 tasks total
  const int qt   = task & 127;
  const int head = (task >> 7) & (NHEAD - 1);
  const int b    = task >> 11;

  const _Float16* Qh = Qb  + ((size_t)(b * NHEAD + head) * T) * HD;
  const _Float16* Kh = Kb  + ((size_t)(b * NHEAD + head) * T) * HD;
  const _Float16* Vh = Vtb + ((size_t)(b * NHEAD + head) * HD) * T;
  const unsigned char* mk = mask + (size_t)b * T;
  _Float16* pw = pbuf + wid * 512;

  const _Float16* qrow = Qh + (size_t)(qt * 16 + r) * HD;
  const v16h qf0 = load_frag_a(qrow, hf);          // d = 0..31
  const v16h qf1 = load_frag_a(qrow + 32, hf);     // d = 32..63

  F8 o[4];
  float mi[8], li[8];
  #pragma unroll
  for (int dt = 0; dt < 4; dt++)
    #pragma unroll
    for (int i = 0; i < 8; i++) o[dt].f[i] = 0.f;
  #pragma unroll
  for (int i = 0; i < 8; i++) { mi[i] = -1e30f; li[i] = 0.f; }

  const float scale = 0.125f;  // hd^-0.5 = 1/8

  for (int kc = 0; kc < T; kc += 32) {
    // ---- scores S = Q @ K^T (two 16-col sub-tiles) ----
    F8 s[2];
    #pragma unroll
    for (int ks = 0; ks < 2; ks++) {
      const _Float16* krow = Kh + (size_t)(kc + ks * 16 + r) * HD;  // B-op: lane r = k-column
      v16h kf0 = load_frag_b(krow, hf);        // d 0..31
      v16h kf1 = load_frag_b(krow + 32, hf);   // d 32..63
      F8 z;
      #pragma unroll
      for (int i = 0; i < 8; i++) z.f[i] = 0.f;
      v8f acc = __builtin_amdgcn_wmma_f32_16x16x32_f16(false, qf0, false, kf0, (short)0, z.v, false, false);
      acc     = __builtin_amdgcn_wmma_f32_16x16x32_f16(false, qf1, false, kf1, (short)0, acc, false, false);
      s[ks].v = acc;
    }

    const float msk0 = mk[kc + r]      ? 0.f : -1e30f;
    const float msk1 = mk[kc + 16 + r] ? 0.f : -1e30f;

    // ---- online softmax; stage P (16x32) into per-wave LDS ----
    #pragma unroll
    for (int i = 0; i < 8; i++) {
      float x0 = s[0].f[i] * scale + msk0;
      float x1 = s[1].f[i] * scale + msk1;
      float mx = wave16_max(fmaxf(x0, x1));
      float mnew = fmaxf(mi[i], mx);
      float corr = __expf(mi[i] - mnew);
      float p0 = __expf(x0 - mnew);
      float p1 = __expf(x1 - mnew);
      float rs = wave16_sum(p0 + p1);
      li[i] = li[i] * corr + rs;
      mi[i] = mnew;
      o[0].f[i] *= corr; o[1].f[i] *= corr; o[2].f[i] *= corr; o[3].f[i] *= corr;
      const int prow = (i + 8 * hf) * 32;      // C-layout row -> LDS row
      pw[prow + r]      = (_Float16)p0;
      pw[prow + 16 + r] = (_Float16)p1;
    }
    // re-gather P as A-operand fragment (same-wave DS ordering)
    v16h pf = load_frag_a(pw + r * 32, hf);

    // ---- O += P @ V  (V^T rows stream contiguously) ----
    #pragma unroll
    for (int dt = 0; dt < 4; dt++) {
      const _Float16* vrow = Vh + (size_t)(dt * 16 + r) * T + kc;   // B-op: lane r = d-column
      v16h vf = load_frag_b(vrow, hf);
      o[dt].v = __builtin_amdgcn_wmma_f32_16x16x32_f16(false, pf, false, vf, (short)0, o[dt].v, false, false);
    }
  }

  // ---- normalize and store [b*T][dm] f16 ----
  #pragma unroll
  for (int i = 0; i < 8; i++) {
    const float inv = li[i] > 0.f ? 1.f / li[i] : 0.f;
    const int qr = qt * 16 + i + 8 * hf;
    const size_t base = ((size_t)b * T + qr) * DMODEL;
    #pragma unroll
    for (int dt = 0; dt < 4; dt++) {
      const int col = head * HD + dt * 16 + r;
      out[base + col] = (_Float16)(o[dt].f[i] * inv);
    }
  }
}

// ---------------------------------------------------------------------------
// y = rmsnorm(query + proj, g_out)
// ---------------------------------------------------------------------------
__global__ void final_rmsnorm_kernel(const float* __restrict__ q,
                                     const float* __restrict__ p,
                                     const float* __restrict__ g,
                                     float* __restrict__ y) {
  const int row = blockIdx.x;
  const float* qr = q + (size_t)row * DMODEL;
  const float* pr = p + (size_t)row * DMODEL;
  const int tid = threadIdx.x;
  float ss = 0.f;
  for (int i = tid; i < DMODEL; i += 256) { float v = qr[i] + pr[i]; ss += v * v; }
  #pragma unroll
  for (int off = 16; off > 0; off >>= 1) ss += __shfl_xor(ss, off, 32);
  __shared__ float red[8];
  if ((tid & 31) == 0) red[tid >> 5] = ss;
  __syncthreads();
  float tot = red[0] + red[1] + red[2] + red[3] + red[4] + red[5] + red[6] + red[7];
  float rinv = rsqrtf(tot * (1.0f / DMODEL) + EPSV);
  float* yr = y + (size_t)row * DMODEL;
  for (int i = tid; i < DMODEL; i += 256) yr[i] = (qr[i] + pr[i]) * rinv * g[i];
}

// ---------------------------------------------------------------------------
extern "C" void kernel_launch(void* const* d_in, const int* in_sizes, int n_in,
                              void* d_out, int out_size, void* d_ws, size_t ws_size,
                              hipStream_t stream) {
  (void)in_sizes; (void)n_in; (void)out_size; (void)ws_size;
  const float* query   = (const float*)d_in[0];
  const float* context = (const float*)d_in[1];
  const unsigned char* cmask = (const unsigned char*)d_in[2];
  const float* Wq  = (const float*)d_in[3];
  const float* Wk  = (const float*)d_in[4];
  const float* Wv  = (const float*)d_in[5];
  const float* Wo  = (const float*)d_in[6];
  const float* g_q = (const float*)d_in[7];
  const float* g_c = (const float*)d_in[8];
  const float* g_o = (const float*)d_in[9];
  float* out = (float*)d_out;

  char* ws = (char*)d_ws;
  const size_t MB = 1024u * 1024u;
  // Workspace layout (48 MB, with dead-buffer overlays):
  _Float16* qn  = (_Float16*)(ws + 0 * MB);    // 8MB  norm(query) f16
  _Float16* cn  = (_Float16*)(ws + 8 * MB);    // 8MB  norm(context) f16
  _Float16* WqT = (_Float16*)(ws + 16 * MB);   // 2MB
  _Float16* WkT = (_Float16*)(ws + 18 * MB);   // 2MB
  _Float16* WvT = (_Float16*)(ws + 20 * MB);   // 2MB
  _Float16* WoT = (_Float16*)(ws + 22 * MB);   // 2MB
  _Float16* Qb  = (_Float16*)(ws + 24 * MB);   // 8MB  [b][h][t][hd]
  _Float16* Kbf = (_Float16*)(ws + 32 * MB);   // 8MB  [b][h][t][hd]
  _Float16* Vtb = (_Float16*)(ws + 40 * MB);   // 8MB  [b][h][hd][t]
  _Float16* ao  = qn;                          // overlay: qn dead after Q proj
  float*    pj  = (float*)(ws + 24 * MB);      // overlay: Qb/Kbf dead after attention

  rmsnorm_cast_kernel<<<MROWS, 256, 0, stream>>>(query, g_q, qn);
  rmsnorm_cast_kernel<<<MROWS, 256, 0, stream>>>(context, g_c, cn);

  dim3 tg(32, 32);
  wtrans_kernel<<<tg, 256, 0, stream>>>(Wq, WqT);
  wtrans_kernel<<<tg, 256, 0, stream>>>(Wk, WkT);
  wtrans_kernel<<<tg, 256, 0, stream>>>(Wv, WvT);
  wtrans_kernel<<<tg, 256, 0, stream>>>(Wo, WoT);

  dim3 gg(DMODEL / 64, MROWS / 128);  // (16, 32)
  gemm_wmma_kernel<<<gg, 256, 0, stream>>>(qn, WqT, Qb,  nullptr, 0);
  gemm_wmma_kernel<<<gg, 256, 0, stream>>>(cn, WkT, Kbf, nullptr, 0);
  gemm_wmma_kernel<<<gg, 256, 0, stream>>>(cn, WvT, Vtb, nullptr, 1);

  flash_attn_kernel<<<(BATCH * NHEAD * (T / 16)) / 8, 256, 0, stream>>>(Qb, Kbf, Vtb, cmask, ao);

  gemm_wmma_kernel<<<gg, 256, 0, stream>>>(ao, WoT, nullptr, pj, 2);

  final_rmsnorm_kernel<<<MROWS, 256, 0, stream>>>(query, pj, g_o, out);
}
